// onlyGAT_51780125720666
// MI455X (gfx1250) — matrix operand order repaired and checked by reading, not compiled, not run on hardware
//
#include <hip/hip_runtime.h>
#include <hip/hip_bf16.h>
#include <float.h>

typedef __attribute__((ext_vector_type(16))) __bf16 v16bf;
typedef __attribute__((ext_vector_type(8)))  __bf16 v8bf;
typedef __attribute__((ext_vector_type(8)))  float  v8f;

#define N_NODES 20000
#define N_EDGES 320000
#define N_GRAPHS 64
#define IN_DIM 512
#define HID 256
#define HEADS 2
#define HH 512            // HEADS*HID
#define NCLS 10
#define EPS_BN 1e-5f
#define SLOPE 0.2f

#define CAT16(lo, hi) __builtin_shufflevector((lo), (hi), 0, 1, 2, 3, 4, 5, 6, 7, \
                                              8, 9, 10, 11, 12, 13, 14, 15)

// ---------------------------------------------------------------- utilities

__device__ __forceinline__ void atomicMaxFloat(float* addr, float v) {
    // monotonic in both int (for v>=0) and reversed-unsigned (for v<0) orderings
    if (v >= 0.f) atomicMax((int*)addr, __float_as_int(v));
    else          atomicMin((unsigned int*)addr, __float_as_uint(v));
}

__global__ void k_fill(float* __restrict__ p, float v, int n) {
    int i = blockIdx.x * blockDim.x + threadIdx.x;
    if (i < n) p[i] = v;
}

__global__ void k_f32_to_bf16(const float* __restrict__ s, __bf16* __restrict__ d, int n) {
    int i = blockIdx.x * blockDim.x + threadIdx.x;
    if (i < n) d[i] = (__bf16)s[i];
}

// ---------------------------------------------------------------- WMMA GEMM
// C[M,Nc] = (accumulate ? C : 0) + A[M,K](bf16) x B[K,Nc](bf16) + bias[Nc]
// Block: 256 threads = 8 waves (4 M x 2 N), block tile 128x64, wave tile 32x32,
// K step 32. B tile staged TRANSPOSED in LDS so all fragment loads are
// 16-byte ds_load_b128; 4 v_wmma per K-step per wave (2 A-frag x 2 B-frag reuse).
__global__ __launch_bounds__(256) void k_gemm_bf16(
    const __bf16* __restrict__ A, const __bf16* __restrict__ B,
    const float* __restrict__ bias, float* __restrict__ C,
    int M, int K, int Nc, int accumulate)
{
    __shared__ __bf16 sA[128][32];    // row-major A tile
    __shared__ __bf16 sBt[64][40];    // B tile transposed [n][k]; 40 keeps rows 16B-aligned

    const int tid  = threadIdx.x;
    const int lane = tid & 31;
    const int wave = tid >> 5;
    const int wm   = wave & 3;     // 0..3 : 32-row M sub-tile
    const int wn   = wave >> 2;    // 0..1 : 32-col N sub-tile
    const int rowBase = blockIdx.x * 128;
    const int colBase = blockIdx.y * 64;

    v8f acc00 = {}, acc01 = {}, acc10 = {}, acc11 = {};

    // cooperative staging coordinates
    const int arow = tid >> 1;          // 0..127
    const int acol = (tid & 1) * 16;    // 0 or 16 (16 bf16 = 32B per thread)
    const int brow = tid >> 3;          // 0..31  (k within tile)
    const int bcol = (tid & 7) * 8;     // 0..56  (n within tile)

    const int fm = lane & 15;
    const int kh = lane >> 4;           // K-half owned by this lane

    for (int k0 = 0; k0 < K; k0 += 32) {
        // ---- stage A tile (row-guarded), two 16B vector stores per thread ----
        {
            const int gr = rowBase + arow;
            if (gr < M) {
                const v8bf* srcp = (const v8bf*)(A + (size_t)gr * K + k0 + acol);
                *(v8bf*)&sA[arow][acol]     = srcp[0];
                *(v8bf*)&sA[arow][acol + 8] = srcp[1];
            } else {
                #pragma unroll
                for (int i = 0; i < 16; ++i) sA[arow][acol + i] = (__bf16)0.f;
            }
        }
        // ---- stage B tile transposed: one 16B global load, 8 b16 LDS stores ----
        {
            const v8bf bv = *(const v8bf*)(B + (size_t)(k0 + brow) * Nc + colBase + bcol);
            #pragma unroll
            for (int i = 0; i < 8; ++i) sBt[bcol + i][brow] = bv[i];
        }
        // prefetch next A tile (global_prefetch_b8 on gfx1250)
        if (k0 + 32 < K) {
            const int gr = rowBase + arow;
            if (gr < M) __builtin_prefetch(A + (size_t)gr * K + k0 + 32 + acol, 0, 1);
        }
        __syncthreads();

        // ---- A fragments: rows wm*32+fm and +16; each half = aligned 16B LDS load ----
        const v8bf* ap0 = (const v8bf*)&sA[wm * 32 + fm][0];
        const v8bf* ap1 = (const v8bf*)&sA[wm * 32 + 16 + fm][0];
        const v16bf a0 = CAT16(ap0[kh], ap0[2 + kh]);   // K: kh*8..+7 , 16+kh*8..+7
        const v16bf a1 = CAT16(ap1[kh], ap1[2 + kh]);

        // ---- B fragments: cols wn*32+fm and +16; K: kh*8..+7 , 16+kh*8..+7 in sBt row ----
        const v8bf* bp0 = (const v8bf*)&sBt[wn * 32 + fm][0];
        const v8bf* bp1 = (const v8bf*)&sBt[wn * 32 + 16 + fm][0];
        const v16bf b0 = CAT16(bp0[kh], bp0[2 + kh]);
        const v16bf b1 = CAT16(bp1[kh], bp1[2 + kh]);

        acc00 = __builtin_amdgcn_wmma_f32_16x16x32_bf16(false, a0, false, b0,
                                                        (short)0, acc00, false, false);
        acc01 = __builtin_amdgcn_wmma_f32_16x16x32_bf16(false, a0, false, b1,
                                                        (short)0, acc01, false, false);
        acc10 = __builtin_amdgcn_wmma_f32_16x16x32_bf16(false, a1, false, b0,
                                                        (short)0, acc10, false, false);
        acc11 = __builtin_amdgcn_wmma_f32_16x16x32_bf16(false, a1, false, b1,
                                                        (short)0, acc11, false, false);
        __syncthreads();
    }

    // ---- store D (VGPR r -> row (lane>>4)*8+r, col lane&15) ----
    const int rb  = (lane >> 4) * 8;
    const int gn0 = colBase + wn * 32 + fm;
    const int gn1 = gn0 + 16;
    #pragma unroll
    for (int r = 0; r < 8; ++r) {
        const int gm0 = rowBase + wm * 32 + rb + r;
        const int gm1 = gm0 + 16;
        if (gm0 < M) {
            const size_t i0 = (size_t)gm0 * Nc + gn0;
            const size_t i1 = (size_t)gm0 * Nc + gn1;
            const float p0 = accumulate ? C[i0] : 0.f;
            const float p1 = accumulate ? C[i1] : 0.f;
            C[i0] = p0 + acc00[r] + bias[gn0];
            C[i1] = p1 + acc01[r] + bias[gn1];
        }
        if (gm1 < M) {
            const size_t i0 = (size_t)gm1 * Nc + gn0;
            const size_t i1 = (size_t)gm1 * Nc + gn1;
            const float p0 = accumulate ? C[i0] : 0.f;
            const float p1 = accumulate ? C[i1] : 0.f;
            C[i0] = p0 + acc10[r] + bias[gn0];
            C[i1] = p1 + acc11[r] + bias[gn1];
        }
    }
}

// ---------------------------------------------------------------- GATv2 edge stages

// one wave per (edge, head): e = sum_d leakyrelu(fs[src]+fd[dst]) * attn ; segment-max by dst
__global__ __launch_bounds__(256) void k_edge_score(
    const float* __restrict__ fs, const float* __restrict__ fd,
    const int* __restrict__ src, const int* __restrict__ dst,
    const float* __restrict__ attn, float* __restrict__ e, float* __restrict__ emax)
{
    const int wid  = (blockIdx.x * blockDim.x + threadIdx.x) >> 5;
    const int lane = threadIdx.x & 31;
    if (wid >= N_EDGES * HEADS) return;
    const int edge = wid >> 1;
    const int h    = wid & 1;
    const int s = src[edge], d = dst[edge];
    const float* ps = fs + (size_t)s * HH + h * HID;
    const float* pd = fd + (size_t)d * HH + h * HID;
    const float* pa = attn + h * HID;
    float acc = 0.f;
    #pragma unroll
    for (int i = 0; i < HID / 32; ++i) {
        const int dim = lane + i * 32;
        float m = ps[dim] + pd[dim];
        m = m > 0.f ? m : SLOPE * m;
        acc += m * pa[dim];
    }
    #pragma unroll
    for (int o = 16; o > 0; o >>= 1) acc += __shfl_xor(acc, o, 32);
    if (lane == 0) {
        e[wid] = acc;
        atomicMaxFloat(&emax[(size_t)d * HEADS + h], acc);
    }
}

__global__ void k_edge_exp(const float* __restrict__ e, const int* __restrict__ dst,
                           const float* __restrict__ emax,
                           float* __restrict__ ee, float* __restrict__ denom)
{
    const int i = blockIdx.x * blockDim.x + threadIdx.x;
    if (i >= N_EDGES * HEADS) return;
    const int edge = i >> 1;
    const int h    = i & 1;
    const int d = dst[edge];
    const float v = expf(e[i] - emax[(size_t)d * HEADS + h]);
    ee[i] = v;
    atomicAdd(&denom[(size_t)d * HEADS + h], v);
}

// one wave per (edge, head): out[dst] += alpha * fs[src]
__global__ __launch_bounds__(256) void k_edge_aggregate(
    const float* __restrict__ fs, const int* __restrict__ src, const int* __restrict__ dst,
    const float* __restrict__ ee, const float* __restrict__ denom, float* __restrict__ out)
{
    const int wid  = (blockIdx.x * blockDim.x + threadIdx.x) >> 5;
    const int lane = threadIdx.x & 31;
    if (wid >= N_EDGES * HEADS) return;
    const int edge = wid >> 1;
    const int h    = wid & 1;
    const int s = src[edge], d = dst[edge];
    const float alpha = ee[wid] / denom[(size_t)d * HEADS + h];
    const float* ps = fs + (size_t)s * HH + h * HID;
    float* po = out + (size_t)d * HH + h * HID;
    #pragma unroll
    for (int i = 0; i < HID / 32; ++i) {
        const int dim = lane + i * 32;
        atomicAdd(&po[dim], alpha * ps[dim]);
    }
}

__global__ void k_elu(float* __restrict__ x, int n) {
    const int i = blockIdx.x * blockDim.x + threadIdx.x;
    if (i < n) {
        const float v = x[i];
        x[i] = v > 0.f ? v : expf(v) - 1.f;
    }
}

// ---------------------------------------------------------------- BatchNorm

__global__ __launch_bounds__(256) void k_bn_stats(const float* __restrict__ h1,
                                                  float* __restrict__ stats)
{
    __shared__ float s1[256];
    __shared__ float s2[256];
    const int c = blockIdx.x;                 // one column per block (HH blocks)
    float a = 0.f, b = 0.f;
    for (int r = threadIdx.x; r < N_NODES; r += 256) {
        const float v = h1[(size_t)r * HH + c];
        a += v; b += v * v;
    }
    s1[threadIdx.x] = a; s2[threadIdx.x] = b;
    __syncthreads();
    for (int o = 128; o > 0; o >>= 1) {
        if (threadIdx.x < o) {
            s1[threadIdx.x] += s1[threadIdx.x + o];
            s2[threadIdx.x] += s2[threadIdx.x + o];
        }
        __syncthreads();
    }
    if (threadIdx.x == 0) { stats[c] = s1[0]; stats[HH + c] = s2[0]; }
}

__global__ void k_bn_apply(const float* __restrict__ h1, const float* __restrict__ stats,
                           const float* __restrict__ gamma, const float* __restrict__ beta,
                           __bf16* __restrict__ h1b, int n)
{
    const int i = blockIdx.x * blockDim.x + threadIdx.x;
    if (i >= n) return;
    const int c = i & (HH - 1);
    const float mu  = stats[c] * (1.f / N_NODES);
    const float var = stats[HH + c] * (1.f / N_NODES) - mu * mu;
    const float v = (h1[i] - mu) * rsqrtf(var + EPS_BN) * gamma[c] + beta[c];
    h1b[i] = (__bf16)v;
}

// ---------------------------------------------------------------- pooling + classifier

__global__ __launch_bounds__(256) void k_gate(const float* __restrict__ hr,
                                              const float* __restrict__ wg,
                                              const float* __restrict__ bg,
                                              const int* __restrict__ gid,
                                              float* __restrict__ gate, float* __restrict__ gmax)
{
    const int wid  = (blockIdx.x * blockDim.x + threadIdx.x) >> 5;
    const int lane = threadIdx.x & 31;
    if (wid >= N_NODES) return;
    const float* p = hr + (size_t)wid * HID;
    float acc = 0.f;
    #pragma unroll
    for (int i = 0; i < HID / 32; ++i) {
        const int dim = lane + i * 32;
        acc += p[dim] * wg[dim];
    }
    #pragma unroll
    for (int o = 16; o > 0; o >>= 1) acc += __shfl_xor(acc, o, 32);
    if (lane == 0) {
        const float g = acc + bg[0];
        gate[wid] = g;
        atomicMaxFloat(&gmax[gid[wid]], g);
    }
}

__global__ void k_gexp(const float* __restrict__ gate, const int* __restrict__ gid,
                       const float* __restrict__ gmax,
                       float* __restrict__ gexp, float* __restrict__ gden)
{
    const int i = blockIdx.x * blockDim.x + threadIdx.x;
    if (i >= N_NODES) return;
    const float v = expf(gate[i] - gmax[gid[i]]);
    gexp[i] = v;
    atomicAdd(&gden[gid[i]], v);
}

__global__ __launch_bounds__(256) void k_pool(const float* __restrict__ hr,
                                              const int* __restrict__ gid,
                                              const float* __restrict__ gexp,
                                              const float* __restrict__ gden,
                                              float* __restrict__ hg)
{
    const int wid  = (blockIdx.x * blockDim.x + threadIdx.x) >> 5;
    const int lane = threadIdx.x & 31;
    if (wid >= N_NODES) return;
    const int g = gid[wid];
    const float a = gexp[wid] / gden[g];
    const float* p = hr + (size_t)wid * HID;
    float* o = hg + (size_t)g * HID;
    #pragma unroll
    for (int i = 0; i < HID / 32; ++i) {
        const int dim = lane + i * 32;
        atomicAdd(&o[dim], a * p[dim]);
    }
}

__global__ __launch_bounds__(128) void k_classifier(const float* __restrict__ hg,
    const float* __restrict__ W1, const float* __restrict__ b1,
    const float* __restrict__ W2, const float* __restrict__ b2, float* __restrict__ out)
{
    __shared__ float t[HID / 2];
    const int g = blockIdx.x, j = threadIdx.x;
    const float* x = hg + (size_t)g * HID;
    float acc = b1[j];
    for (int k = 0; k < HID; ++k) acc += x[k] * W1[(size_t)k * (HID / 2) + j];
    t[j] = acc > 0.f ? acc : 0.f;
    __syncthreads();
    if (j < NCLS) {
        float a2 = b2[j];
        for (int k = 0; k < HID / 2; ++k) a2 += t[k] * W2[(size_t)k * NCLS + j];
        out[(size_t)g * NCLS + j] = a2;
    }
}

// ---------------------------------------------------------------- launch

extern "C" void kernel_launch(void* const* d_in, const int* in_sizes, int n_in,
                              void* d_out, int out_size, void* d_ws, size_t ws_size,
                              hipStream_t stream) {
    const float* feat   = (const float*)d_in[0];
    const int*   src    = (const int*)d_in[1];
    const int*   dst    = (const int*)d_in[2];
    const int*   gid    = (const int*)d_in[3];
    const float* W_src  = (const float*)d_in[4];
    const float* b_src  = (const float*)d_in[5];
    const float* W_dst  = (const float*)d_in[6];
    const float* b_dst  = (const float*)d_in[7];
    const float* attn   = (const float*)d_in[8];
    const float* W_skip = (const float*)d_in[9];
    const float* b_skip = (const float*)d_in[10];
    const float* gamma  = (const float*)d_in[11];
    const float* beta   = (const float*)d_in[12];
    const float* W_red  = (const float*)d_in[13];
    const float* b_red  = (const float*)d_in[14];
    const float* w_gate = (const float*)d_in[15];
    const float* b_gate = (const float*)d_in[16];
    const float* W1     = (const float*)d_in[17];
    const float* b1     = (const float*)d_in[18];
    const float* W2     = (const float*)d_in[19];
    const float* b2     = (const float*)d_in[20];
    float* out = (float*)d_out;

    // bump allocator on d_ws (256B aligned slices)
    char* wsb = (char*)d_ws;
    size_t off = 0;
    auto alloc = [&](size_t bytes) -> void* {
        void* p = wsb + off;
        off += (bytes + 255) & ~(size_t)255;
        return p;
    };
    __bf16* featb  = (__bf16*)alloc((size_t)N_NODES * IN_DIM * 2);
    __bf16* wsrcb  = (__bf16*)alloc((size_t)IN_DIM * HH * 2);
    __bf16* wdstb  = (__bf16*)alloc((size_t)IN_DIM * HH * 2);
    __bf16* wskipb = (__bf16*)alloc((size_t)IN_DIM * HH * 2);
    __bf16* wredb  = (__bf16*)alloc((size_t)HH * HID * 2);
    float*  fs     = (float*)alloc((size_t)N_NODES * HH * 4);
    float*  fd     = (float*)alloc((size_t)N_NODES * HH * 4);
    float*  h1     = (float*)alloc((size_t)N_NODES * HH * 4);   // gat out -> elu -> +skip
    float*  e      = (float*)alloc((size_t)N_EDGES * HEADS * 4);
    float*  ee     = (float*)alloc((size_t)N_EDGES * HEADS * 4);
    float*  emax   = (float*)alloc((size_t)N_NODES * HEADS * 4);
    float*  denom  = (float*)alloc((size_t)N_NODES * HEADS * 4);
    float*  stats  = (float*)alloc((size_t)2 * HH * 4);
    __bf16* h1b    = (__bf16*)alloc((size_t)N_NODES * HH * 2);
    float*  hr     = (float*)alloc((size_t)N_NODES * HID * 4);
    float*  gate   = (float*)alloc((size_t)N_NODES * 4);
    float*  gexp   = (float*)alloc((size_t)N_NODES * 4);
    float*  gmax   = (float*)alloc((size_t)N_GRAPHS * 4);
    float*  gden   = (float*)alloc((size_t)N_GRAPHS * 4);
    float*  hg     = (float*)alloc((size_t)N_GRAPHS * HID * 4);
    (void)ws_size; (void)in_sizes; (void)n_in; (void)out_size;

    auto cdiv = [](int a, int b) { return (a + b - 1) / b; };
    const float NEG = -3.402823466e38f;

    // bf16 conversions
    k_f32_to_bf16<<<cdiv(N_NODES * IN_DIM, 256), 256, 0, stream>>>(feat, featb, N_NODES * IN_DIM);
    k_f32_to_bf16<<<cdiv(IN_DIM * HH, 256), 256, 0, stream>>>(W_src, wsrcb, IN_DIM * HH);
    k_f32_to_bf16<<<cdiv(IN_DIM * HH, 256), 256, 0, stream>>>(W_dst, wdstb, IN_DIM * HH);
    k_f32_to_bf16<<<cdiv(IN_DIM * HH, 256), 256, 0, stream>>>(W_skip, wskipb, IN_DIM * HH);
    k_f32_to_bf16<<<cdiv(HH * HID, 256), 256, 0, stream>>>(W_red, wredb, HH * HID);

    // init accumulators / segment buffers (every call -> deterministic)
    k_fill<<<cdiv(N_NODES * HH, 256), 256, 0, stream>>>(h1, 0.f, N_NODES * HH);
    k_fill<<<cdiv(N_NODES * HEADS, 256), 256, 0, stream>>>(emax, NEG, N_NODES * HEADS);
    k_fill<<<cdiv(N_NODES * HEADS, 256), 256, 0, stream>>>(denom, 0.f, N_NODES * HEADS);
    k_fill<<<1, 256, 0, stream>>>(gmax, NEG, N_GRAPHS);
    k_fill<<<1, 256, 0, stream>>>(gden, 0.f, N_GRAPHS);
    k_fill<<<cdiv(N_GRAPHS * HID, 256), 256, 0, stream>>>(hg, 0.f, N_GRAPHS * HID);

    // fs / fd projections (WMMA)
    dim3 gproj(cdiv(N_NODES, 128), HH / 64);
    k_gemm_bf16<<<gproj, 256, 0, stream>>>(featb, wsrcb, b_src, fs, N_NODES, IN_DIM, HH, 0);
    k_gemm_bf16<<<gproj, 256, 0, stream>>>(featb, wdstb, b_dst, fd, N_NODES, IN_DIM, HH, 0);

    // edge softmax + aggregation (wave per (edge, head))
    const int nwaves_e = N_EDGES * HEADS;
    k_edge_score<<<cdiv(nwaves_e * 32, 256), 256, 0, stream>>>(fs, fd, src, dst, attn, e, emax);
    k_edge_exp<<<cdiv(nwaves_e, 256), 256, 0, stream>>>(e, dst, emax, ee, denom);
    k_edge_aggregate<<<cdiv(nwaves_e * 32, 256), 256, 0, stream>>>(fs, src, dst, ee, denom, h1);

    // h1 = elu(gat) + skip   (skip GEMM accumulates into h1)
    k_elu<<<cdiv(N_NODES * HH, 256), 256, 0, stream>>>(h1, N_NODES * HH);
    k_gemm_bf16<<<gproj, 256, 0, stream>>>(featb, wskipb, b_skip, h1, N_NODES, IN_DIM, HH, 1);

    // BatchNorm (training-mode batch stats), emit bf16 activations
    k_bn_stats<<<HH, 256, 0, stream>>>(h1, stats);
    k_bn_apply<<<cdiv(N_NODES * HH, 256), 256, 0, stream>>>(h1, stats, gamma, beta, h1b,
                                                            N_NODES * HH);

    // hr = bn(h1) @ W_red + b_red (WMMA)
    dim3 gred(cdiv(N_NODES, 128), HID / 64);
    k_gemm_bf16<<<gred, 256, 0, stream>>>(h1b, wredb, b_red, hr, N_NODES, HH, HID, 0);

    // global attention pooling over graphs
    k_gate<<<cdiv(N_NODES * 32, 256), 256, 0, stream>>>(hr, w_gate, b_gate, gid, gate, gmax);
    k_gexp<<<cdiv(N_NODES, 256), 256, 0, stream>>>(gate, gid, gmax, gexp, gden);
    k_pool<<<cdiv(N_NODES * 32, 256), 256, 0, stream>>>(hr, gid, gexp, gden, hg);

    // classifier MLP -> out[64,10]
    k_classifier<<<N_GRAPHS, 128, 0, stream>>>(hg, W1, b1, W2, b2, out);
}
